// Attention_63513976373985
// MI455X (gfx1250) — compile-verified, matrix operand order checked by reading, not run on hardware
//
#include <hip/hip_runtime.h>
#include <hip/hip_bf16.h>

// ---------------------------------------------------------------------------
// Additive multi-head attention on MI455X (gfx1250, wave32, WMMA).
// - All GEMMs: v_wmma_f32_16x16x32_bf16 with split-bf16 (hi+lo) operands,
//   3 WMMAs per K=32 step for ~f32 accuracy on the bf16 matrix pipe.
// - Hot per-head GEMM: k-outer, 16 accumulator tiles per wave (A read once),
//   B chunk staged into LDS with gfx1250 async-to-LDS copies, double-buffered,
//   shared by all 8 waves of the workgroup.
// ---------------------------------------------------------------------------

#define HNUM 8
#define FDIM 512
#define CDIM 256
#define SDIM 512
#define BDIM 64
#define M2   (SDIM * BDIM)   // 32768 flattened d2 rows
#define K2C  (2 * CDIM)      // 512

typedef __bf16 bf16;
typedef __attribute__((ext_vector_type(16))) bf16  v16bf;
typedef __attribute__((ext_vector_type(8)))  float v8f;
typedef __attribute__((ext_vector_type(4)))  int   v4i;

#if defined(__gfx1250__) && __has_builtin(__builtin_amdgcn_global_load_async_to_lds_b128) && __has_builtin(__builtin_amdgcn_s_wait_asynccnt)
#define HAVE_ASYNC 1
#else
#define HAVE_ASYNC 0
#endif

// 16-byte global -> LDS copy (async on gfx1250; sync fallback elsewhere)
static __device__ __forceinline__ void cp16_g2l(const void* g, void* l) {
#if HAVE_ASYNC
    // builtin prototype (from clang): (v4i addrspace(1)*, v4i addrspace(3)*, imm, imm)
    // integer round-trip: generic->AS1 keeps the 64-bit address; generic->AS3
    // is "low 32 bits" on AMDGPU, which is exactly the LDS byte offset.
    __builtin_amdgcn_global_load_async_to_lds_b128(
        (__attribute__((address_space(1))) v4i*)(unsigned long long)g,
        (__attribute__((address_space(3))) v4i*)(unsigned int)(unsigned long long)l,
        0, 0);
#else
    *(int4*)l = *(const int4*)g;
#endif
}
static __device__ __forceinline__ void wait_async_all() {
#if HAVE_ASYNC
    __builtin_amdgcn_s_wait_asynccnt(0);
#endif
}

static __device__ __forceinline__ v8f wmma_bf(v16bf a, v16bf b, v8f c) {
    return __builtin_amdgcn_wmma_f32_16x16x32_bf16(
        /*neg_a=*/false, a, /*neg_b=*/false, b,
        /*c_mod=*/(short)0, c, /*reuse_a=*/false, /*reuse_b=*/false);
}

// split-bf16 product: (ah+al)*(bh+bl) ~= ah*bh + al*bh + ah*bl
static __device__ __forceinline__ v8f wmma3(v16bf ah, v16bf al,
                                            v16bf bh, v16bf bl, v8f c) {
    c = wmma_bf(ah, bh, c);
    c = wmma_bf(al, bh, c);
    c = wmma_bf(ah, bl, c);
    return c;
}

static __device__ __forceinline__ v16bf ldv(const bf16* p) {
    return *(const v16bf*)p;
}

static __device__ __forceinline__ void splitw(float x, bf16* hp, bf16* lp) {
    bf16 h = (bf16)x;
    *hp = h;
    *lp = (bf16)(x - (float)h);
}

// ---------------------------------------------------------------------------
// Conversion / packing kernels
// ---------------------------------------------------------------------------

// f32 row-major -> split bf16 hi/lo, same layout (A operands)
__global__ void k_split(const float* __restrict__ src,
                        bf16* __restrict__ hi, bf16* __restrict__ lo, int n) {
    int i = blockIdx.x * blockDim.x + threadIdx.x;
    if (i < n) splitw(src[i], hi + i, lo + i);
}

// f32 [K,N] row-major -> WMMA-B packed split bf16:
// p(k,n) = ((k/32)*N + n)*32 + ((k>>4)&1)*16 + (k&15)
// (for fixed k-chunk of 32, the 32*N elements are CONTIGUOUS -> async-friendly)
__global__ void k_packB(const float* __restrict__ src,
                        bf16* __restrict__ hi, bf16* __restrict__ lo,
                        int K, int N, int total) {
    int i = blockIdx.x * blockDim.x + threadIdx.x;
    if (i >= total) return;
    int mk  = K * N;
    int mat = i / mk, rem = i - mat * mk;
    int k   = rem / N, n  = rem - k * N;
    int p   = mat * mk + ((k >> 5) * N + n) * 32 + ((k >> 4) & 1) * 16 + (k & 15);
    splitw(src[i], hi + p, lo + p);
}

// ---------------------------------------------------------------------------
// Stage 1: d4 = relu(d1 @ w1 + b1), stored as split bf16 [64,256]
// ---------------------------------------------------------------------------
__global__ void k_d4(const bf16* __restrict__ d1h, const bf16* __restrict__ d1l,
                     const bf16* __restrict__ w1ph, const bf16* __restrict__ w1pl,
                     const float* __restrict__ b1,
                     bf16* __restrict__ d4h, bf16* __restrict__ d4l) {
    int lane = threadIdx.x & 31, kh = lane >> 4, lm = lane & 15;
    int wid  = blockIdx.x * (blockDim.x >> 5) + (threadIdx.x >> 5);
    int mt = wid >> 4, nt = wid & 15;
    int row0 = mt * 16, n0 = nt * 16;
    v8f acc = {};
    for (int k0 = 0; k0 < FDIM; k0 += 32) {
        int ai = (row0 + lm) * FDIM + k0 + kh * 16;
        int bi = ((k0 >> 5) * CDIM + n0 + lm) * 32 + kh * 16;
        acc = wmma3(ldv(d1h + ai), ldv(d1l + ai),
                    ldv(w1ph + bi), ldv(w1pl + bi), acc);
    }
    float bias = b1[n0 + lm];
#pragma unroll
    for (int v = 0; v < 8; v++) {
        int m = v + 8 * kh, r = row0 + m;
        float x = fmaxf(acc[v] + bias, 0.f);
        splitw(x, d4h + r * CDIM + n0 + lm, d4l + r * CDIM + n0 + lm);
    }
}

// ---------------------------------------------------------------------------
// Stage 2 (fused): d3 = relu(d2 @ w1 + b1) -> split bf16 [32768,256]
//                  tv = tanh(d2 @ wv)      -> per-batch WMMA-B packed split
// One wave computes a 16x64 tile for BOTH outputs; d2 is read once.
// ---------------------------------------------------------------------------
__global__ void k_d3tv(const bf16* __restrict__ d2h, const bf16* __restrict__ d2l,
                       const bf16* __restrict__ w1ph, const bf16* __restrict__ w1pl,
                       const bf16* __restrict__ wvph, const bf16* __restrict__ wvpl,
                       const float* __restrict__ b1,
                       bf16* __restrict__ d3h, bf16* __restrict__ d3l,
                       bf16* __restrict__ tvph, bf16* __restrict__ tvpl) {
    int lane = threadIdx.x & 31, kh = lane >> 4, lm = lane & 15;
    int wid  = blockIdx.x * (blockDim.x >> 5) + (threadIdx.x >> 5);
    int rt = wid >> 2, ng = wid & 3;
    int row0 = rt * 16, n0b = ng * 64;
    v8f acc1[4] = {};
    v8f acc2[4] = {};
    for (int k0 = 0; k0 < FDIM; k0 += 32) {
        int ai = (row0 + lm) * FDIM + k0 + kh * 16;
        v16bf ah = ldv(d2h + ai), al = ldv(d2l + ai);
#pragma unroll
        for (int j = 0; j < 4; j++) {
            int bi = ((k0 >> 5) * CDIM + n0b + j * 16 + lm) * 32 + kh * 16;
            acc1[j] = wmma3(ah, al, ldv(w1ph + bi), ldv(w1pl + bi), acc1[j]);
            acc2[j] = wmma3(ah, al, ldv(wvph + bi), ldv(wvpl + bi), acc2[j]);
        }
    }
#pragma unroll
    for (int j = 0; j < 4; j++) {
        int n = n0b + j * 16 + lm;
        float bias = b1[n];
#pragma unroll
        for (int v = 0; v < 8; v++) {
            int m = v + 8 * kh, r = row0 + m;
            float x = fmaxf(acc1[j][v] + bias, 0.f);
            splitw(x, d3h + r * CDIM + n, d3l + r * CDIM + n);
            float t = tanhf(acc2[j][v]);
            int s = r >> 6, b = r & 63;   // r = s*B + b
            int pi = b * (SDIM * CDIM)
                   + ((s >> 5) * CDIM + n) * 32 + ((s >> 4) & 1) * 16 + (s & 15);
            splitw(t, tvph + pi, tvpl + pi);
        }
    }
}

// ---------------------------------------------------------------------------
// Stage 3: per head h: us = tanh(d5 @ W_h); atts[h,b,s] = us . P_h
// d5 row r = [d3[r,:], d4[r&63,:]] (concat) — never materialized.
// Block = 8 waves, one head, 8 consecutive 16-row tiles.
// k-outer loop: each wave holds all 16 accumulator tiles (full C=256) so A
// is read exactly once; the 32KB B k-chunk (contiguous in packed layout) is
// async-copied into double-buffered LDS and shared by all 8 waves, with the
// copy of chunk k+1 overlapping the 48 WMMAs of chunk k.
// ---------------------------------------------------------------------------
__global__ void k_heads(const bf16* __restrict__ d3h, const bf16* __restrict__ d3l,
                        const bf16* __restrict__ d4h, const bf16* __restrict__ d4l,
                        const bf16* __restrict__ Wph, const bf16* __restrict__ Wpl,
                        const float* __restrict__ P, float* __restrict__ atts) {
    __shared__ __align__(128) char smem[2 * 32768];   // 2 bufs x (16KB hi + 16KB lo)
    int lane = threadIdx.x & 31, kh = lane >> 4, lm = lane & 15;
    int wave = threadIdx.x >> 5;
    int t    = threadIdx.x;
    int h    = blockIdx.y;
    int rt   = blockIdx.x * (blockDim.x >> 5) + wave;
    int row0 = rt * 16;
    int r    = row0 + lm;          // this lane's A row
    int bidx = r & 63;
    const bf16* gBh = Wph + h * (K2C * CDIM);  // packed: kblk chunks of 8192 el
    const bf16* gBl = Wpl + h * (K2C * CDIM);

    // cooperative stage of one 32-k chunk (hi 16KB + lo 16KB) into buf
    auto stage = [&](int buf, int kblk) {
        const char* gh = (const char*)(gBh + kblk * 8192) + t * 64;
        const char* gl = (const char*)(gBl + kblk * 8192) + t * 64;
        char* lh = smem + buf * 32768 + t * 64;
        char* ll = smem + buf * 32768 + 16384 + t * 64;
#pragma unroll
        for (int j = 0; j < 4; j++) {
            cp16_g2l(gh + j * 16, lh + j * 16);
            cp16_g2l(gl + j * 16, ll + j * 16);
        }
    };

    v8f acc[16] = {};
    stage(0, 0);
    for (int kblk = 0; kblk < 16; kblk++) {
        int buf = kblk & 1;
        wait_async_all();        // this thread's pending copies are in LDS
        __syncthreads();         // everyone's copies visible; prev buf free
        if (kblk < 15) stage(buf ^ 1, kblk + 1);   // overlap with compute

        int kseg = kblk * 32 + kh * 16;
        const bf16 *pah, *pal;
        if (kseg < CDIM) { pah = d3h + r * CDIM + kseg;
                           pal = d3l + r * CDIM + kseg; }
        else             { pah = d4h + bidx * CDIM + (kseg - CDIM);
                           pal = d4l + bidx * CDIM + (kseg - CDIM); }
        v16bf ah = ldv(pah), al = ldv(pal);

        const bf16* lbh = (const bf16*)(smem + buf * 32768);
        const bf16* lbl = (const bf16*)(smem + buf * 32768 + 16384);
#pragma unroll
        for (int nt = 0; nt < 16; nt++) {
            int off = (nt * 16 + lm) * 32 + kh * 16;
            acc[nt] = wmma3(ah, al, ldv(lbh + off), ldv(lbl + off), acc[nt]);
        }
    }

    // fused epilogue: tanh, dot with P over C, tree-reduce over the 16 n-lanes
    float pacc[8] = {0.f, 0.f, 0.f, 0.f, 0.f, 0.f, 0.f, 0.f};
#pragma unroll
    for (int nt = 0; nt < 16; nt++) {
        float pv = P[h * CDIM + nt * 16 + lm];
#pragma unroll
        for (int v = 0; v < 8; v++) pacc[v] += tanhf(acc[nt][v]) * pv;
    }
#pragma unroll
    for (int v = 0; v < 8; v++) {
        float x = pacc[v];
        x += __shfl_xor(x, 1, 32);
        x += __shfl_xor(x, 2, 32);
        x += __shfl_xor(x, 4, 32);
        x += __shfl_xor(x, 8, 32);
        pacc[v] = x;
    }
    if (lm == 0) {
#pragma unroll
        for (int v = 0; v < 8; v++) {
            int m = v + 8 * kh, rr = row0 + m;
            int s = rr >> 6, b = rr & 63;
            atts[(h * BDIM + b) * SDIM + s] = pacc[v];
        }
    }
}

// ---------------------------------------------------------------------------
// Stage 4: softmax over S per (h,b); write scores as split bf16 A-rows.
// ---------------------------------------------------------------------------
__global__ void k_softmax(const float* __restrict__ atts,
                          bf16* __restrict__ scoresh, bf16* __restrict__ scoresl) {
    __shared__ float red[256];
    int row = blockIdx.x;                // h*64 + b
    int t = threadIdx.x;
    const float* a = atts + row * SDIM;
    float x0 = a[t], x1 = a[t + 256];
    red[t] = fmaxf(x0, x1);
    __syncthreads();
    for (int s = 128; s > 0; s >>= 1) {
        if (t < s) red[t] = fmaxf(red[t], red[t + s]);
        __syncthreads();
    }
    float m = red[0];
    __syncthreads();
    float e0 = __expf(x0 - m), e1 = __expf(x1 - m);
    red[t] = e0 + e1;
    __syncthreads();
    for (int s = 128; s > 0; s >>= 1) {
        if (t < s) red[t] += red[t + s];
        __syncthreads();
    }
    float inv = 1.f / red[0];
    splitw(e0 * inv, scoresh + row * SDIM + t,       scoresl + row * SDIM + t);
    splitw(e1 * inv, scoresh + row * SDIM + t + 256, scoresl + row * SDIM + t + 256);
}

// ---------------------------------------------------------------------------
// Stage 5: per batch b: vs[h,c] = sum_s scores[h,b,s] * tv[b,s,c]
// A = scores (M=16, rows >= HNUM zeroed in-register), B = packed tv[b].
// ---------------------------------------------------------------------------
__global__ void k_vs(const bf16* __restrict__ scoresh, const bf16* __restrict__ scoresl,
                     const bf16* __restrict__ tvph, const bf16* __restrict__ tvpl,
                     bf16* __restrict__ vsh, bf16* __restrict__ vsl) {
    int lane = threadIdx.x & 31, kh = lane >> 4, lm = lane & 15;
    int wid  = blockIdx.x * (blockDim.x >> 5) + (threadIdx.x >> 5);
    int b = wid >> 2, ng = wid & 3;
    int n0b = ng * 64;
    v8f acc[4] = {};
    for (int k0 = 0; k0 < SDIM; k0 += 32) {
        v16bf ah = {}, al = {};
        if (lm < HNUM) {
            int ai = (lm * BDIM + b) * SDIM + k0 + kh * 16;
            ah = ldv(scoresh + ai);
            al = ldv(scoresl + ai);
        }
#pragma unroll
        for (int j = 0; j < 4; j++) {
            int bi = b * (SDIM * CDIM)
                   + ((k0 >> 5) * CDIM + n0b + j * 16 + lm) * 32 + kh * 16;
            acc[j] = wmma3(ah, al, ldv(tvph + bi), ldv(tvpl + bi), acc[j]);
        }
    }
#pragma unroll
    for (int j = 0; j < 4; j++) {
        int n = n0b + j * 16 + lm;
#pragma unroll
        for (int v = 0; v < 8; v++) {
            int m = v + 8 * kh;
            if (m < HNUM) {
                int oi = b * (HNUM * CDIM) + m * CDIM + n;
                splitw(acc[j][v], vsh + oi, vsl + oi);
            }
        }
    }
}

// ---------------------------------------------------------------------------
// Stage 6: out = relu(vs @ wcc + bcc), [64,2048]@[2048,128] -> [64,128] f32
// ---------------------------------------------------------------------------
__global__ void k_out(const bf16* __restrict__ vsh, const bf16* __restrict__ vsl,
                      const bf16* __restrict__ wccph, const bf16* __restrict__ wccpl,
                      const float* __restrict__ bcc, float* __restrict__ out) {
    int lane = threadIdx.x & 31, kh = lane >> 4, lm = lane & 15;
    int wid  = blockIdx.x * (blockDim.x >> 5) + (threadIdx.x >> 5);
    int mt = wid >> 3, nt = wid & 7;
    int row0 = mt * 16, n0 = nt * 16;
    v8f acc = {};
    for (int k0 = 0; k0 < HNUM * CDIM; k0 += 32) {
        int ai = (row0 + lm) * (HNUM * CDIM) + k0 + kh * 16;
        int bi = ((k0 >> 5) * 128 + n0 + lm) * 32 + kh * 16;
        acc = wmma3(ldv(vsh + ai), ldv(vsl + ai),
                    ldv(wccph + bi), ldv(wccpl + bi), acc);
    }
    float bias = bcc[n0 + lm];
#pragma unroll
    for (int v = 0; v < 8; v++) {
        int m = v + 8 * kh;
        out[(row0 + m) * 128 + n0 + lm] = fmaxf(acc[v] + bias, 0.f);
    }
}

// ---------------------------------------------------------------------------
// Host launcher
// ---------------------------------------------------------------------------
extern "C" void kernel_launch(void* const* d_in, const int* in_sizes, int n_in,
                              void* d_out, int out_size, void* d_ws, size_t ws_size,
                              hipStream_t stream) {
    const float* d1  = (const float*)d_in[0];   // [64,512]
    const float* d2  = (const float*)d_in[1];   // [512,64,512]
    const float* w1  = (const float*)d_in[2];   // [512,256]
    const float* b1  = (const float*)d_in[3];   // [256]
    const float* W   = (const float*)d_in[4];   // [8,512,256]
    const float* P   = (const float*)d_in[5];   // [8,256]
    const float* wv  = (const float*)d_in[6];   // [512,256]
    const float* wcc = (const float*)d_in[7];   // [2048,128]
    const float* bcc = (const float*)d_in[8];   // [128]
    float* out = (float*)d_out;                 // [64,128]

    // bump allocator over workspace (256B aligned carve-outs)
    char* wp = (char*)d_ws;
    auto alloc = [&](size_t bytes) -> char* {
        char* r = wp;
        wp += (bytes + 255) & ~(size_t)255;
        return r;
    };
    bf16* d2h   = (bf16*)alloc((size_t)M2 * FDIM * 2);
    bf16* d2l   = (bf16*)alloc((size_t)M2 * FDIM * 2);
    bf16* d1h   = (bf16*)alloc((size_t)BDIM * FDIM * 2);
    bf16* d1l   = (bf16*)alloc((size_t)BDIM * FDIM * 2);
    bf16* w1ph  = (bf16*)alloc((size_t)FDIM * CDIM * 2);
    bf16* w1pl  = (bf16*)alloc((size_t)FDIM * CDIM * 2);
    bf16* wvph  = (bf16*)alloc((size_t)FDIM * CDIM * 2);
    bf16* wvpl  = (bf16*)alloc((size_t)FDIM * CDIM * 2);
    bf16* Wph   = (bf16*)alloc((size_t)HNUM * K2C * CDIM * 2);
    bf16* Wpl   = (bf16*)alloc((size_t)HNUM * K2C * CDIM * 2);
    bf16* wccph = (bf16*)alloc((size_t)HNUM * CDIM * 128 * 2);
    bf16* wccpl = (bf16*)alloc((size_t)HNUM * CDIM * 128 * 2);
    bf16* d4h   = (bf16*)alloc((size_t)BDIM * CDIM * 2);
    bf16* d4l   = (bf16*)alloc((size_t)BDIM * CDIM * 2);
    bf16* d3h   = (bf16*)alloc((size_t)M2 * CDIM * 2);
    bf16* d3l   = (bf16*)alloc((size_t)M2 * CDIM * 2);
    bf16* tvph  = (bf16*)alloc((size_t)BDIM * SDIM * CDIM * 2);
    bf16* tvpl  = (bf16*)alloc((size_t)BDIM * SDIM * CDIM * 2);
    float* atts = (float*)alloc((size_t)HNUM * BDIM * SDIM * 4);
    bf16* sch   = (bf16*)alloc((size_t)HNUM * BDIM * SDIM * 2);
    bf16* scl   = (bf16*)alloc((size_t)HNUM * BDIM * SDIM * 2);
    bf16* vsh   = (bf16*)alloc((size_t)BDIM * HNUM * CDIM * 2);
    bf16* vsl   = (bf16*)alloc((size_t)BDIM * HNUM * CDIM * 2);

    const int T = 256;

    // --- convert / pack ---
    {
        int n = M2 * FDIM;                       // 16,777,216
        k_split<<<dim3(n / T), dim3(T), 0, stream>>>(d2, d2h, d2l, n);
    }
    {
        int n = BDIM * FDIM;                     // 32,768
        k_split<<<dim3(n / T), dim3(T), 0, stream>>>(d1, d1h, d1l, n);
    }
    {
        int n = FDIM * CDIM;                     // 131,072
        k_packB<<<dim3(n / T), dim3(T), 0, stream>>>(w1, w1ph, w1pl, FDIM, CDIM, n);
        k_packB<<<dim3(n / T), dim3(T), 0, stream>>>(wv, wvph, wvpl, FDIM, CDIM, n);
    }
    {
        int n = HNUM * K2C * CDIM;               // 1,048,576
        k_packB<<<dim3(n / T), dim3(T), 0, stream>>>(W, Wph, Wpl, K2C, CDIM, n);
    }
    {
        int n = HNUM * CDIM * 128;               // 262,144
        k_packB<<<dim3(n / T), dim3(T), 0, stream>>>(wcc, wccph, wccpl, HNUM * CDIM, 128, n);
    }

    // --- stage 1: d4 --- 64 waves (4 m-tiles x 16 n-tiles)
    k_d4<<<dim3(8), dim3(T), 0, stream>>>(d1h, d1l, w1ph, w1pl, b1, d4h, d4l);

    // --- stage 2: d3 + tv fused --- 2048 row-tiles x 4 n-groups = 8192 waves
    k_d3tv<<<dim3(1024), dim3(T), 0, stream>>>(d2h, d2l, w1ph, w1pl, wvph, wvpl,
                                               b1, d3h, d3l, tvph, tvpl);

    // --- stage 3: per-head GEMM + P-dot --- (2048/8 row-groups) x 8 heads
    k_heads<<<dim3(256, 8), dim3(T), 0, stream>>>(d3h, d3l, d4h, d4l,
                                                  Wph, Wpl, P, atts);

    // --- stage 4: softmax over S --- one block per (h,b)
    k_softmax<<<dim3(HNUM * BDIM), dim3(T), 0, stream>>>(atts, sch, scl);

    // --- stage 5: scores @ tv per batch --- 64 b x 4 n-groups = 256 waves
    k_vs<<<dim3(32), dim3(T), 0, stream>>>(sch, scl, tvph, tvpl, vsh, vsl);

    // --- stage 6: final projection --- 4 m-tiles x 8 n-tiles = 32 waves
    k_out<<<dim3(4), dim3(T), 0, stream>>>(vsh, vsl, wccph, wccpl, bcc, out);
}
